// Attention_72559177499201
// MI455X (gfx1250) — matrix operand order repaired and checked by reading, not compiled, not run on hardware
//
#include <hip/hip_runtime.h>

// ---------------------------------------------------------------------------
// MI455X (gfx1250). Compute-bound problem (~344 GFLOP vs ~0.4GB traffic), so
// everything heavy runs on v_wmma_f32_16x16x32_bf16 (f32 accumulate).
// Pipeline:
//   K0: cast q_bar, a_bar -> bf16 (row-major)
//   K0b: transpose-cast Wg -> WgT (bf16), a_bar -> a_barT (bf16)
//        => every GEMM B-operand is N-major: LDS staging is pure 16B copies
//           via global_load_async_to_lds_b128 (no transpose bank conflicts).
//   K1: transform = q@Wg + bg            -> bf16 (WMMA, fused bias+cast)
//   K2: scoresT[b,a,q] = a_bar@transform^T -> f32 (WMMA)
//   K3: row softmax over q, in place (f32)
//   K4: h = g @ a_bar -> f32 d_out       (WMMA, A=f32 g converted on load)
// GEMM: 128x128x64 tiles, 8 waves x (64x32), double-buffered LDS with async
// memory->LDS fills (ASYNCcnt) overlapping the WMMA stage.
// ---------------------------------------------------------------------------

typedef __bf16 bf16;
typedef __attribute__((ext_vector_type(8)))  bf16  v8bf;
typedef __attribute__((ext_vector_type(16))) bf16  v16bf;
typedef __attribute__((ext_vector_type(8)))  float v8f;
typedef __attribute__((ext_vector_type(4)))  float v4f;

constexpr int BM = 128, BN = 128, BK = 64;   // workgroup tile
constexpr int LDK = BK + 8;                  // LDS row pitch (16B aligned)
constexpr int WM = 64, WN = 32;              // per-wave tile (2x4 wave grid)
constexpr int MFRAG = WM / 16, NFRAG = WN / 16;

__device__ inline v16bf cat8(v8bf lo, v8bf hi) {
  v16bf r;
#pragma unroll
  for (int i = 0; i < 8; ++i) { r[i] = lo[i]; r[i + 8] = hi[i]; }
  return r;
}

// Async 16B copy: global memory -> LDS, tracked by ASYNCcnt (per-wave).
__device__ inline void async_b128(unsigned lds_off, const void* g) {
  asm volatile("global_load_async_to_lds_b128 %0, %1, off"
               :: "v"(lds_off), "v"((unsigned long long)(uintptr_t)g)
               : "memory");
}
template <int N> __device__ inline void wait_async() {
  if constexpr (N == 0)      asm volatile("s_wait_asynccnt 0x0" ::: "memory");
  else if constexpr (N == 4) asm volatile("s_wait_asynccnt 0x4" ::: "memory");
  else                       asm volatile("s_wait_asynccnt 0x8" ::: "memory");
}

// C(MxN, f32 acc) = A(MxK) * B^T-given  [+bias]
//  A_IS_F32 : A is f32 in global; software-pipelined VGPR convert to bf16
//  B is ALWAYS N-major (N rows x K cols row-major) -> async 16B staging
template <bool A_IS_F32, bool OUT_BF16, bool BIAS>
__global__ __launch_bounds__(256) void gemm_wmma_bf16(
    const void* __restrict__ Ag_, const bf16* __restrict__ Bg_,
    void* __restrict__ Cg_, const float* __restrict__ bias,
    int M, int N, int K, int lda, int ldb, int ldc,
    long sA, long sB, long sC) {
  __shared__ bf16 As[2][BM][LDK];   // 2 x 18KB
  __shared__ bf16 Bs[2][BN][LDK];   // 2 x 18KB  (total 72KB of 320KB WGP LDS)

  const int t    = threadIdx.x;
  const int lane = t & 31;
  const int wave = t >> 5;
  const int wm   = (wave >> 2) * WM;
  const int wn   = (wave & 3) * WN;
  const int m0   = blockIdx.y * BM;
  const int n0   = blockIdx.x * BN;
  const int b    = blockIdx.z;

  const bf16*  Ab = A_IS_F32 ? nullptr : (const bf16*)Ag_ + (long)b * sA;
  const float* Af = A_IS_F32 ? (const float*)Ag_ + (long)b * sA : nullptr;
  const bf16*  Bg = Bg_ + (long)b * sB;

  constexpr int NA = A_IS_F32 ? 4 : 8;  // async ops in flight per stage
  v8f acc[MFRAG][NFRAG] = {};
  v4f ar[4][2];                         // manual-path staging regs (A_IS_F32)

  // ---- stage issue helpers --------------------------------------------------
  auto issue_async = [&](int kt, int nb) {
    const int k0 = kt * BK;
    if (!A_IS_F32) {
#pragma unroll
      for (int it = 0; it < 4; ++it) {
        int chunk = t + it * 256;                 // 1024 chunks of 8 bf16
        int m = chunk >> 3, k8 = (chunk & 7) << 3;
        async_b128((unsigned)(uintptr_t)&As[nb][m][k8],
                   Ab + (long)(m0 + m) * lda + k0 + k8);
      }
    }
#pragma unroll
    for (int it = 0; it < 4; ++it) {
      int chunk = t + it * 256;
      int n = chunk >> 3, k8 = (chunk & 7) << 3;
      async_b128((unsigned)(uintptr_t)&Bs[nb][n][k8],
                 Bg + (long)(n0 + n) * ldb + k0 + k8);
    }
  };
  auto manload = [&](int kt) {                    // A_IS_F32: global -> regs
    const int k0 = kt * BK;
#pragma unroll
    for (int it = 0; it < 4; ++it) {
      int chunk = t + it * 256;
      int m = chunk >> 3, k8 = (chunk & 7) << 3;
      const float* src = Af + (long)(m0 + m) * lda + k0 + k8;
      ar[it][0] = *(const v4f*)(src);
      ar[it][1] = *(const v4f*)(src + 4);
    }
  };
  auto manstore = [&](int nb) {                   // regs -> bf16 -> LDS
#pragma unroll
    for (int it = 0; it < 4; ++it) {
      int chunk = t + it * 256;
      int m = chunk >> 3, k8 = (chunk & 7) << 3;
      v8bf o;
#pragma unroll
      for (int i = 0; i < 4; ++i) { o[i] = (bf16)ar[it][0][i]; o[i + 4] = (bf16)ar[it][1][i]; }
      *(v8bf*)&As[nb][m][k8] = o;
    }
  };

  // ---- software pipeline ----------------------------------------------------
  const int KT = K / BK;
  issue_async(0, 0);
  if (A_IS_F32) { manload(0); manstore(0); }

  for (int kt = 0; kt < KT; ++kt) {
    const int cur = kt & 1, nxt = cur ^ 1;
    if (kt + 1 < KT) {
      issue_async(kt + 1, nxt);                   // overlap with this stage
      if (A_IS_F32) manload(kt + 1);
      wait_async<NA>();                           // current stage landed
    } else {
      wait_async<0>();
    }
    __syncthreads();

#pragma unroll
    for (int kk = 0; kk < BK; kk += 32) {
      v16bf afr[MFRAG], bfr[NFRAG];
      // A 16x32 bf16: lanes0-15 row M=l, K={kk..+7, kk+16..+23}; lanes16-31 +8.
      const int arow = wm + (lane & 15);
      const int akb  = kk + ((lane >> 4) << 3);
#pragma unroll
      for (int i = 0; i < MFRAG; ++i) {
        const bf16* p = &As[cur][arow + i * 16][akb];
        afr[i] = cat8(*(const v8bf*)p, *(const v8bf*)(p + 16));
      }
      // B 32x16 bf16: lane n=l&15; lanes0-15 K={kk..+7, kk+16..+23}; lanes16-31 +8.
      const int brow = wn + (lane & 15);
      const int bks  = kk + ((lane >> 4) << 3);
#pragma unroll
      for (int j = 0; j < NFRAG; ++j) {
        const bf16* p = &Bs[cur][brow + j * 16][bks];
        bfr[j] = cat8(*(const v8bf*)p, *(const v8bf*)(p + 16));
      }
#pragma unroll
      for (int i = 0; i < MFRAG; ++i)
#pragma unroll
        for (int j = 0; j < NFRAG; ++j)
          acc[i][j] = __builtin_amdgcn_wmma_f32_16x16x32_bf16(
              false, afr[i], false, bfr[j], (short)0, acc[i][j], false, false);
    }

    if (A_IS_F32 && kt + 1 < KT) manstore(nxt);   // after compute, before barrier
    __syncthreads();
  }

  // ---- epilogue: C layout VGPR r, lanes0-15 row=r, lanes16-31 row=r+8 ------
  const int ln = lane & 15, lh = lane >> 4;
  bf16*  Cb = (bf16*)Cg_ + (long)b * sC;
  float* Cf = (float*)Cg_ + (long)b * sC;
#pragma unroll
  for (int i = 0; i < MFRAG; ++i) {
#pragma unroll
    for (int j = 0; j < NFRAG; ++j) {
      const int col  = n0 + wn + j * 16 + ln;
      const float bv = BIAS ? bias[col] : 0.0f;
#pragma unroll
      for (int r = 0; r < 8; ++r) {
        const int row = m0 + wm + i * 16 + r + lh * 8;
        const float v = acc[i][j][r] + bv;
        if (OUT_BF16) Cb[(long)row * ldc + col] = (bf16)v;
        else          Cf[(long)row * ldc + col] = v;
      }
    }
  }
}

// Row softmax over q (2048 contiguous f32 per row), in place.
__global__ __launch_bounds__(256) void softmax_q_rows(float* __restrict__ sc) {
  __shared__ float red[256];
  const int t = threadIdx.x;
  float* row = sc + (long)blockIdx.x * 2048;
  float x[8];
#pragma unroll
  for (int i = 0; i < 8; ++i) x[i] = row[t + i * 256];
  float m = x[0];
#pragma unroll
  for (int i = 1; i < 8; ++i) m = fmaxf(m, x[i]);
  red[t] = m; __syncthreads();
  for (int s = 128; s > 0; s >>= 1) {
    if (t < s) red[t] = fmaxf(red[t], red[t + s]);
    __syncthreads();
  }
  m = red[0]; __syncthreads();
  float e[8], sum = 0.f;
#pragma unroll
  for (int i = 0; i < 8; ++i) { e[i] = __expf(x[i] - m); sum += e[i]; }
  red[t] = sum; __syncthreads();
  for (int s = 128; s > 0; s >>= 1) {
    if (t < s) red[t] += red[t + s];
    __syncthreads();
  }
  const float inv = 1.0f / red[0];
#pragma unroll
  for (int i = 0; i < 8; ++i) row[t + i * 256] = e[i] * inv;
}

__global__ __launch_bounds__(256) void cast_f32_to_bf16(
    const float* __restrict__ in, bf16* __restrict__ out, long n8) {
  long i = (long)blockIdx.x * 256 + threadIdx.x;
  if (i >= n8) return;
  v4f x0 = ((const v4f*)in)[2 * i];
  v4f x1 = ((const v4f*)in)[2 * i + 1];
  v8bf o;
#pragma unroll
  for (int k = 0; k < 4; ++k) { o[k] = (bf16)x0[k]; o[k + 4] = (bf16)x1[k]; }
  ((v8bf*)out)[i] = o;
}

// out[c][r] = (bf16) in[r][c]; in: R x C f32 row-major. 32x32 LDS tiles.
__global__ __launch_bounds__(256) void transpose_cast_f32_bf16(
    const float* __restrict__ in, bf16* __restrict__ out,
    int R, int C, long sIn, long sOut) {
  __shared__ float tile[32][33];
  const int tx = threadIdx.x & 31;
  const int ty = threadIdx.x >> 5;             // 0..7
  const float* ib = in  + (long)blockIdx.z * sIn;
  bf16*        ob = out + (long)blockIdx.z * sOut;
  const int c0 = blockIdx.x * 32, r0 = blockIdx.y * 32;
#pragma unroll
  for (int i = 0; i < 4; ++i)
    tile[ty + 8 * i][tx] = ib[(long)(r0 + ty + 8 * i) * C + c0 + tx];
  __syncthreads();
#pragma unroll
  for (int i = 0; i < 4; ++i)
    ob[(long)(c0 + ty + 8 * i) * R + r0 + tx] = (bf16)tile[tx][ty + 8 * i];
}

extern "C" void kernel_launch(void* const* d_in, const int* in_sizes, int n_in,
                              void* d_out, int out_size, void* d_ws, size_t ws_size,
                              hipStream_t stream) {
  constexpr int B = 16, S = 2048, H = 1024;
  const float* q  = (const float*)d_in[0];
  const float* a  = (const float*)d_in[1];
  const float* Wg = (const float*)d_in[2];
  const float* bg = (const float*)d_in[3];

  // Workspace (514 MB):
  //  qb 64MB | ab 64MB | wbT 2MB | tb 64MB | abT 64MB | scoresT 256MB (f32)
  bf16*  qb  = (bf16*)d_ws;
  bf16*  ab  = qb  + (size_t)B * S * H;
  bf16*  wbT = ab  + (size_t)B * S * H;
  bf16*  tb  = wbT + (size_t)H * H;
  bf16*  abT = tb  + (size_t)B * S * H;
  float* sc  = (float*)(abT + (size_t)B * S * H);

  {
    long n8 = (long)B * S * H / 8;
    cast_f32_to_bf16<<<dim3((unsigned)(n8 / 256)), 256, 0, stream>>>(q, qb, n8);
    cast_f32_to_bf16<<<dim3((unsigned)(n8 / 256)), 256, 0, stream>>>(a, ab, n8);
  }
  // WgT[n][k] = Wg[k][n]  (N-major B for GEMM1)
  transpose_cast_f32_bf16<<<dim3(H / 32, H / 32, 1), 256, 0, stream>>>(
      Wg, wbT, H, H, 0, 0);
  // abT[b][h][q] = a_bar[b][q][h]  (N-major B for GEMM3)
  transpose_cast_f32_bf16<<<dim3(H / 32, S / 32, B), 256, 0, stream>>>(
      a, abT, S, H, (long)S * H, (long)S * H);

  // K1: transform = q @ Wg + bg   (M=B*S, N=H, K=H) -> bf16 tb
  gemm_wmma_bf16<false, true, true>
      <<<dim3(H / BN, (B * S) / BM, 1), 256, 0, stream>>>(
          qb, wbT, tb, bg, B * S, H, H, H, H, H, 0, 0, 0);

  // K2: scoresT[b] = a_bar[b] @ transform[b]^T (M=S(a), N=S(q), K=H) -> f32
  //     B-operand = tb (q,h) is already N-major.
  gemm_wmma_bf16<false, false, false>
      <<<dim3(S / BN, S / BM, B), 256, 0, stream>>>(
          ab, tb, sc, nullptr, S, S, H, H, H, S,
          (long)S * H, (long)S * H, (long)S * S);

  // K3: softmax over q (contiguous rows of scoresT), in place
  softmax_q_rows<<<dim3(B * S), 256, 0, stream>>>(sc);

  // K4: h[b] = g[b] @ a_bar[b]  (M=S(a), N=H, K=S(q)); A=f32 g, B=abT N-major
  gemm_wmma_bf16<true, false, false>
      <<<dim3(H / BN, S / BM, B), 256, 0, stream>>>(
          sc, abT, (float*)d_out, nullptr, S, H, S, S, S, H,
          (long)S * S, (long)S * H, (long)S * H);
}
